// GraphTransformerBlock_44040594653886
// MI455X (gfx1250) — compile-verified, matrix operand order checked by reading, not run on hardware
//
#include <hip/hip_runtime.h>

typedef __attribute__((ext_vector_type(16))) _Float16 v16h;
typedef __attribute__((ext_vector_type(8)))  float    v8f;
typedef __attribute__((ext_vector_type(4)))  _Float16 h4;

#define D_MODEL 128
#define NHEAD   8
#define DHEAD   16
#define FF_DIM  256

// ---------------------------------------------------------------- utilities

static __device__ __forceinline__ float warp_sum32(float v) {
#pragma unroll
  for (int m = 16; m >= 1; m >>= 1) v += __shfl_xor(v, m, 32);
  return v;
}

static __device__ __forceinline__ void atomicMaxF(float* addr, float val) {
  // sign-aware int trick; valid for any mix of signs, init must be a float
  if (val >= 0.0f) atomicMax((int*)addr, __float_as_int(val));
  else             atomicMin((unsigned int*)addr, __float_as_uint(val));
}

static __device__ __forceinline__ float gelu_tanh(float x) {
  float x3 = x * x * x;
  return 0.5f * x * (1.0f + tanhf(0.7978845608028654f * (x + 0.044715f * x3)));
}

__global__ void fill_f32_kernel(float* __restrict__ p, float v, long long n) {
  long long i = (long long)blockIdx.x * blockDim.x + threadIdx.x;
  if (i < n) p[i] = v;
}

__global__ void f32_to_f16_kernel(const float* __restrict__ s,
                                  _Float16* __restrict__ d, long long n) {
  long long i = (long long)blockIdx.x * blockDim.x + threadIdx.x;
  if (i < n) d[i] = (_Float16)s[i];
}

// dst[n*K + k] = (f16) src[k*Ncols + n]   (transpose + convert, tiny matrices)
__global__ void wtrans_f16_kernel(const float* __restrict__ src,
                                  _Float16* __restrict__ dst, int K, int Ncols) {
  int i = blockIdx.x * blockDim.x + threadIdx.x;
  if (i >= K * Ncols) return;
  int k = i / Ncols, n = i - k * Ncols;
  dst[(size_t)n * K + k] = (_Float16)src[i];
}

// ---------------------------------------------------------------- LayerNorm

// one wave per row (D=128: each lane owns 4 contiguous floats)
__global__ void ln_to_f16_kernel(const float* __restrict__ x,
                                 const float* __restrict__ g,
                                 const float* __restrict__ b,
                                 _Float16* __restrict__ out, int N) {
  int row  = blockIdx.x * 4 + (threadIdx.x >> 5);
  int lane = threadIdx.x & 31;
  if (row >= N) return;
  const float4 xv = *(const float4*)(x + (size_t)row * D_MODEL + lane * 4);
  float s  = xv.x + xv.y + xv.z + xv.w;
  float s2 = xv.x * xv.x + xv.y * xv.y + xv.z * xv.z + xv.w * xv.w;
  s = warp_sum32(s);  s2 = warp_sum32(s2);
  float mean = s * (1.0f / D_MODEL);
  float var  = s2 * (1.0f / D_MODEL) - mean * mean;
  float inv  = rsqrtf(var + 1e-5f);
  float4 gv = *(const float4*)(g + lane * 4);
  float4 bv = *(const float4*)(b + lane * 4);
  h4 o;
  o.x = (_Float16)((xv.x - mean) * inv * gv.x + bv.x);
  o.y = (_Float16)((xv.y - mean) * inv * gv.y + bv.y);
  o.z = (_Float16)((xv.z - mean) * inv * gv.z + bv.z);
  o.w = (_Float16)((xv.w - mean) * inv * gv.w + bv.w);
  *(h4*)(out + (size_t)row * D_MODEL + lane * 4) = o;
}

// ---------------------------------------------------------------- WMMA GEMM

union V16U { v16h v; uint4 q[2]; };

// D = A(MxK,f16) @ Bt^T (Bt is Ncols x K, f16) + bias; optional gelu, residual.
// One wave computes a 16x16 tile via v_wmma_f32_16x16x32_f16 (K/32 steps).
__global__ void gemm16_wmma_kernel(const _Float16* __restrict__ A, int Kd,
                                   const _Float16* __restrict__ Bt,
                                   const float* __restrict__ bias,
                                   const float* __restrict__ residual,
                                   float* __restrict__ outF,
                                   _Float16* __restrict__ outH,
                                   int M, int Ncols, int act_gelu) {
  int wave = threadIdx.x >> 5;
  int lane = threadIdx.x & 31;
  int m0 = blockIdx.x * 16;
  int n0 = (blockIdx.y * 4 + wave) * 16;
  if (n0 >= Ncols || m0 >= M) return;            // wave-uniform exit

  int l15  = lane & 15;
  int half = lane >> 4;                          // 0: K 0-7/16-23, 1: K 8-15/24-31
  int rowA = m0 + l15; if (rowA > M - 1) rowA = M - 1;   // clamp (no divergence)
  int col  = n0 + l15;

  const _Float16* ap = A  + (size_t)rowA * Kd + half * 8;
  const _Float16* bp = Bt + (size_t)col  * Kd + half * 8;

  float bval = bias[col];
  v8f c;
#pragma unroll
  for (int i = 0; i < 8; ++i) c[i] = bval;

  for (int k0 = 0; k0 < Kd; k0 += 32) {
    V16U a, bb;
    a.q[0]  = *(const uint4*)(ap + k0);
    a.q[1]  = *(const uint4*)(ap + k0 + 16);
    bb.q[0] = *(const uint4*)(bp + k0);
    bb.q[1] = *(const uint4*)(bp + k0 + 16);
    c = __builtin_amdgcn_wmma_f32_16x16x32_f16(
        false, a.v, false, bb.v, (short)0, c, false, false);
  }

#pragma unroll
  for (int j = 0; j < 8; ++j) {
    int r = m0 + j + half * 8;
    if (r >= M) continue;
    float val = c[j];
    if (act_gelu) val = gelu_tanh(val);
    size_t idx = (size_t)r * Ncols + col;
    if (residual) val += residual[idx];
    if (outF) outF[idx] = val;
    if (outH) outH[idx] = (_Float16)val;
  }
}

// ---------------------------------------------------------------- edge phase

// per-edge scores (8 heads) + segment max into mbuf
__global__ void scores_kernel(const float* __restrict__ q,
                              const float* __restrict__ k,
                              const int* __restrict__ edge_index,
                              const int* __restrict__ edge_type,
                              const unsigned char* __restrict__ edge_mask,
                              const float* __restrict__ edge_bias,
                              float* __restrict__ scb,
                              float* __restrict__ mbuf, int E, int T) {
  __shared__ float sbias[64];
  int tb = T * NHEAD;
  if (threadIdx.x < 64) sbias[threadIdx.x] = (threadIdx.x < tb) ? edge_bias[threadIdx.x] : 0.0f;
  __syncthreads();

  int e = blockIdx.x * blockDim.x + threadIdx.x;
  if (e >= E) return;
  int src = edge_index[e];
  int dst = edge_index[E + e];
  int t   = edge_type[e];
  bool ok = edge_mask[e] != 0;

  const float4* qp = (const float4*)(q + (size_t)dst * D_MODEL);
  const float4* kp = (const float4*)(k + (size_t)src * D_MODEL);
  float sc[NHEAD];
#pragma unroll
  for (int h = 0; h < NHEAD; ++h) {
    float acc = 0.0f;
#pragma unroll
    for (int j = 0; j < 4; ++j) {
      float4 a = qp[h * 4 + j];
      float4 b = kp[h * 4 + j];
      acc += a.x * b.x + a.y * b.y + a.z * b.z + a.w * b.w;
    }
    float s = acc * 0.25f + sbias[t * NHEAD + h];   // dh^-0.5 = 0.25
    if (!ok) s = -1e9f;
    sc[h] = s;
    atomicMaxF(&mbuf[(size_t)dst * NHEAD + h], s);
  }
  float4* op = (float4*)(scb + (size_t)e * NHEAD);
  op[0] = make_float4(sc[0], sc[1], sc[2], sc[3]);
  op[1] = make_float4(sc[4], sc[5], sc[6], sc[7]);
}

// p = exp(score - m[dst]); segment-sum into denom; p overwrites scb
__global__ void exp_kernel(float* __restrict__ scb,
                           const float* __restrict__ mbuf,
                           const int* __restrict__ edge_index,
                           float* __restrict__ denom, int E) {
  int e = blockIdx.x * blockDim.x + threadIdx.x;
  if (e >= E) return;
  int dst = edge_index[E + e];
  float4* sp = (float4*)(scb + (size_t)e * NHEAD);
  const float4* mp = (const float4*)(mbuf + (size_t)dst * NHEAD);
  float4 s0 = sp[0], s1 = sp[1], m0 = mp[0], m1 = mp[1];
  float p[NHEAD];
  p[0] = __expf(s0.x - m0.x); p[1] = __expf(s0.y - m0.y);
  p[2] = __expf(s0.z - m0.z); p[3] = __expf(s0.w - m0.w);
  p[4] = __expf(s1.x - m1.x); p[5] = __expf(s1.y - m1.y);
  p[6] = __expf(s1.z - m1.z); p[7] = __expf(s1.w - m1.w);
  sp[0] = make_float4(p[0], p[1], p[2], p[3]);
  sp[1] = make_float4(p[4], p[5], p[6], p[7]);
#pragma unroll
  for (int h = 0; h < NHEAD; ++h)
    atomicAdd(&denom[(size_t)dst * NHEAD + h], p[h]);
}

// msg[dst, h*16 + :] += alpha * v[src, h*16 + :]  (one thread per edge-head)
__global__ void aggregate_kernel(const float* __restrict__ pbuf,
                                 const float* __restrict__ denom,
                                 const float* __restrict__ v,
                                 const int* __restrict__ edge_index,
                                 float* __restrict__ msg, int E) {
  int idx = blockIdx.x * blockDim.x + threadIdx.x;
  if (idx >= E * NHEAD) return;
  int e = idx >> 3, h = idx & 7;
  int src = edge_index[e];
  int dst = edge_index[E + e];
  float p   = pbuf[(size_t)e * NHEAD + h];
  float den = denom[(size_t)dst * NHEAD + h];
  float alpha = p / (den + 1e-9f);
  const float4* vp = (const float4*)(v + (size_t)src * D_MODEL + h * DHEAD);
  float* mp = msg + (size_t)dst * D_MODEL + h * DHEAD;
#pragma unroll
  for (int j = 0; j < 4; ++j) {
    float4 vv = vp[j];
    atomicAdd(mp + j * 4 + 0, alpha * vv.x);
    atomicAdd(mp + j * 4 + 1, alpha * vv.y);
    atomicAdd(mp + j * 4 + 2, alpha * vv.z);
    atomicAdd(mp + j * 4 + 3, alpha * vv.w);
  }
}

// ---------------------------------------------------------------- launcher

extern "C" void kernel_launch(void* const* d_in, const int* in_sizes, int n_in,
                              void* d_out, int out_size, void* d_ws, size_t ws_size,
                              hipStream_t stream) {
  const float* x          = (const float*)d_in[0];
  const int*   edge_index = (const int*)d_in[1];
  const int*   edge_type  = (const int*)d_in[2];
  const unsigned char* edge_mask = (const unsigned char*)d_in[4];
  const float* ln1_g = (const float*)d_in[5];
  const float* ln1_b = (const float*)d_in[6];
  const float* wq = (const float*)d_in[7];  const float* bq = (const float*)d_in[8];
  const float* wk = (const float*)d_in[9];  const float* bk = (const float*)d_in[10];
  const float* wv = (const float*)d_in[11]; const float* bv = (const float*)d_in[12];
  const float* edge_bias = (const float*)d_in[13];
  const float* wo = (const float*)d_in[14]; const float* bo = (const float*)d_in[15];
  const float* ln2_g = (const float*)d_in[16];
  const float* ln2_b = (const float*)d_in[17];
  const float* w1 = (const float*)d_in[18]; const float* b1 = (const float*)d_in[19];
  const float* w2 = (const float*)d_in[20]; const float* b2 = (const float*)d_in[21];

  const int N = in_sizes[0] / D_MODEL;
  const int E = in_sizes[1] / 2;
  const int T = in_sizes[13] / NHEAD;

  // ---- workspace carve-out
  char* p = (char*)d_ws;
  auto take = [&](size_t bytes) -> char* {
    char* r = p; p += (bytes + 255) & ~(size_t)255; return r;
  };
  float*    qb   = (float*)take((size_t)N * D_MODEL * 4);
  float*    kb   = (float*)take((size_t)N * D_MODEL * 4);
  float*    vb   = (float*)take((size_t)N * D_MODEL * 4);
  float*    scb  = (float*)take((size_t)E * NHEAD * 4);
  float*    mbuf = (float*)take((size_t)N * NHEAD * 4);
  float*    den  = (float*)take((size_t)N * NHEAD * 4);
  float*    msg  = (float*)take((size_t)N * D_MODEL * 4);
  float*    x1   = (float*)take((size_t)N * D_MODEL * 4);
  _Float16* hH   = (_Float16*)take((size_t)N * D_MODEL * 2);  // reused 3x
  _Float16* ffH  = (_Float16*)take((size_t)N * FF_DIM * 2);
  _Float16* wqT  = (_Float16*)take((size_t)D_MODEL * D_MODEL * 2);
  _Float16* wkT  = (_Float16*)take((size_t)D_MODEL * D_MODEL * 2);
  _Float16* wvT  = (_Float16*)take((size_t)D_MODEL * D_MODEL * 2);
  _Float16* woT  = (_Float16*)take((size_t)D_MODEL * D_MODEL * 2);
  _Float16* w1T  = (_Float16*)take((size_t)D_MODEL * FF_DIM * 2);
  _Float16* w2T  = (_Float16*)take((size_t)FF_DIM * D_MODEL * 2);

  // ---- weight convert + transpose (tiny)
  {
    int nDD = D_MODEL * D_MODEL, nDF = D_MODEL * FF_DIM;
    wtrans_f16_kernel<<<(nDD + 255) / 256, 256, 0, stream>>>(wq, wqT, D_MODEL, D_MODEL);
    wtrans_f16_kernel<<<(nDD + 255) / 256, 256, 0, stream>>>(wk, wkT, D_MODEL, D_MODEL);
    wtrans_f16_kernel<<<(nDD + 255) / 256, 256, 0, stream>>>(wv, wvT, D_MODEL, D_MODEL);
    wtrans_f16_kernel<<<(nDD + 255) / 256, 256, 0, stream>>>(wo, woT, D_MODEL, D_MODEL);
    wtrans_f16_kernel<<<(nDF + 255) / 256, 256, 0, stream>>>(w1, w1T, D_MODEL, FF_DIM);
    wtrans_f16_kernel<<<(nDF + 255) / 256, 256, 0, stream>>>(w2, w2T, FF_DIM, D_MODEL);
  }

  // ---- init accumulators
  {
    long long nh = (long long)N * NHEAD, nd = (long long)N * D_MODEL;
    fill_f32_kernel<<<(int)((nh + 255) / 256), 256, 0, stream>>>(mbuf, -1e30f, nh);
    fill_f32_kernel<<<(int)((nh + 255) / 256), 256, 0, stream>>>(den, 0.0f, nh);
    fill_f32_kernel<<<(int)((nd + 255) / 256), 256, 0, stream>>>(msg, 0.0f, nd);
  }

  // ---- LN1 -> h (f16)
  ln_to_f16_kernel<<<(N + 3) / 4, 128, 0, stream>>>(x, ln1_g, ln1_b, hH, N);

  // ---- QKV projections (WMMA)
  dim3 gD((N + 15) / 16, D_MODEL / 64);   // 16-row tiles, 4 col-tiles/block
  gemm16_wmma_kernel<<<gD, 128, 0, stream>>>(hH, D_MODEL, wqT, bq, nullptr, qb, nullptr, N, D_MODEL, 0);
  gemm16_wmma_kernel<<<gD, 128, 0, stream>>>(hH, D_MODEL, wkT, bk, nullptr, kb, nullptr, N, D_MODEL, 0);
  gemm16_wmma_kernel<<<gD, 128, 0, stream>>>(hH, D_MODEL, wvT, bv, nullptr, vb, nullptr, N, D_MODEL, 0);

  // ---- edge attention
  scores_kernel<<<(E + 255) / 256, 256, 0, stream>>>(qb, kb, edge_index, edge_type,
                                                     edge_mask, edge_bias, scb, mbuf, E, T);
  exp_kernel<<<(E + 255) / 256, 256, 0, stream>>>(scb, mbuf, edge_index, den, E);
  aggregate_kernel<<<(E * NHEAD + 255) / 256, 256, 0, stream>>>(scb, den, vb, edge_index, msg, E);

  // ---- out projection + residual 1
  f32_to_f16_kernel<<<(int)(((long long)N * D_MODEL + 255) / 256), 256, 0, stream>>>(
      msg, hH, (long long)N * D_MODEL);
  gemm16_wmma_kernel<<<gD, 128, 0, stream>>>(hH, D_MODEL, woT, bo, x, x1, nullptr, N, D_MODEL, 0);

  // ---- LN2 + FFN + residual 2
  ln_to_f16_kernel<<<(N + 3) / 4, 128, 0, stream>>>(x1, ln2_g, ln2_b, hH, N);
  dim3 gF((N + 15) / 16, FF_DIM / 64);
  gemm16_wmma_kernel<<<gF, 128, 0, stream>>>(hH, D_MODEL, w1T, b1, nullptr, nullptr, ffH, N, FF_DIM, 1);
  gemm16_wmma_kernel<<<gD, 128, 0, stream>>>(ffH, FF_DIM, w2T, b2, x1, (float*)d_out, nullptr, N, D_MODEL, 0);
}
